// FineRegister_35966056137509
// MI455X (gfx1250) — matrix operand order repaired and checked by reading, not compiled, not run on hardware
//
#include <hip/hip_runtime.h>

typedef __attribute__((ext_vector_type(16))) _Float16 v16h;
typedef __attribute__((ext_vector_type(8)))  float    v8f;

#define BB 8
#define NN 1536
#define PP 256
#define NEG_INF (-3.0e38f)
#define EPSV 1e-8f
#define BNEPS 1e-5f

// ---------------- input / workspace pointer bundles ----------------
struct In {
  const float *src,*tgt,*srcF,*tgtF;
  // params flattened in jax pytree order (sorted dict keys, tuples in order):
  const float *sfW1,*sfW2,*sfB2,*sfG,*sfBb,*sfM,*sfV;
  const float *sgW1,*sgW2,*sgW3,*sgB3;
  const float *sg1G,*sg1B,*sg1M,*sg1V,*sg2G,*sg2B,*sg2M,*sg2V;
  const float *smW1,*smW2,*smB2,*smG,*smBb,*smM,*smV;
  const float *wfW1,*wfW2,*wfB2,*wfG,*wfBb,*wfM,*wfV;
};

struct Ws {
  float *sig;            // [2][B][N]
  int   *idxS,*idxT;     // [B][P]
  float *srcSel,*tgtSel,*srcCur; // [B][3][P]
  float *A2,*C2,*rowmax; // [B][32][P]
  float *wts;            // [B][P]
  int   *corr;           // [B][P]
  float *Racc,*tacc;     // [B][9], [B][3]
  float *sigW1,*sigB1,*sigW2,*sigB2,*sigW3,*sigB3;
  float *sfWa,*sfWb,*sfWg,*sfSh;
  float *smW1f,*smSh,*smW2v,*smB2v;
  float *wfW1f,*wfSh,*wfW2v,*wfB2v;
};

// ---------------- fold batchnorms into weights ----------------
__global__ __launch_bounds__(256) void k_prep(In in, Ws w) {
  int t = threadIdx.x;
  for (int o = t; o < 64; o += 256) {
    float sc = in.sg1G[o] * rsqrtf(in.sg1V[o] + BNEPS);
    w.sigB1[o] = in.sg1B[o] - in.sg1M[o] * sc;
    for (int i = 0; i < 64; ++i) w.sigW1[o*64+i] = in.sgW1[o*64+i] * sc;
  }
  for (int o = t; o < 32; o += 256) {
    float sc = in.sg2G[o] * rsqrtf(in.sg2V[o] + BNEPS);
    w.sigB2[o] = in.sg2B[o] - in.sg2M[o] * sc;
    for (int i = 0; i < 64; ++i) w.sigW2[o*64+i] = in.sgW2[o*64+i] * sc;
    w.sigW3[o] = in.sgW3[o];

    float scf = in.sfG[o] * rsqrtf(in.sfV[o] + BNEPS);
    w.sfSh[o] = in.sfBb[o] - in.sfM[o] * scf;
    for (int i = 0; i < 64; ++i) {
      w.sfWa[o*64+i] = in.sfW1[o*132+i]    * scf;
      w.sfWb[o*64+i] = in.sfW1[o*132+64+i] * scf;
    }
    for (int i = 0; i < 4; ++i) w.sfWg[o*4+i] = in.sfW1[o*132+128+i] * scf;

    float scm = in.smG[o] * rsqrtf(in.smV[o] + BNEPS);
    w.smSh[o] = in.smBb[o] - in.smM[o] * scm;
    for (int i = 0; i < 32; ++i) w.smW1f[o*32+i] = in.smW1[o*32+i] * scm;
    w.smW2v[o] = in.smW2[o];

    float scw = in.wfG[o] * rsqrtf(in.wfV[o] + BNEPS);
    w.wfSh[o] = in.wfBb[o] - in.wfM[o] * scw;
    for (int i = 0; i < 32; ++i) w.wfW1f[o*32+i] = in.wfW1[o*32+i] * scw;
    w.wfW2v[o] = in.wfW2[o];
  }
  if (t == 0) { w.sigB3[0]=in.sgB3[0]; w.smB2v[0]=in.smB2[0]; w.wfB2v[0]=in.wfB2[0]; }
}

// ---------------- significance MLP (64->64->32->1), both clouds ----------------
__global__ __launch_bounds__(256) void k_sig(In in, Ws w) {
  __shared__ float sW1[64*64];
  __shared__ float sW2[32*64];
  __shared__ float sb1[64], sb2[32], sW3[32];
  int t = threadIdx.x;
  for (int k=t;k<64*64;k+=256) sW1[k]=w.sigW1[k];
  for (int k=t;k<32*64;k+=256) sW2[k]=w.sigW2[k];
  if (t<64) sb1[t]=w.sigB1[t];
  if (t<32) { sb2[t]=w.sigB2[t]; sW3[t]=w.sigW3[t]; }
  __syncthreads();
  long gid = (long)blockIdx.x*256 + t;       // < 2*B*N
  int which = (int)(gid / (BB*NN));
  int rem   = (int)(gid % (BB*NN));
  int b = rem / NN, n = rem % NN;
  const float* F = which ? in.tgtF : in.srcF;
  float x[64];
  #pragma unroll 8
  for (int c=0;c<64;++c) x[c] = F[((long)b*64+c)*NN + n];
  float acc2[32];
  #pragma unroll
  for (int p=0;p<32;++p) acc2[p] = sb2[p];
  for (int o=0;o<64;++o) {
    float a = sb1[o]; const float* wr=&sW1[o*64];
    #pragma unroll 8
    for (int c=0;c<64;++c) a += wr[c]*x[c];
    a = fmaxf(a, 0.f);
    #pragma unroll
    for (int p=0;p<32;++p) acc2[p] += sW2[p*64+o]*a;
  }
  float acc = w.sigB3[0];
  #pragma unroll
  for (int p=0;p<32;++p) acc += sW3[p]*fmaxf(acc2[p], 0.f);
  w.sig[gid] = acc;
}

// ---------------- top-k via bitonic sort (order irrelevant downstream) ----------------
__global__ __launch_bounds__(256) void k_topk(In in, Ws w) {
  __shared__ float v[2048];
  __shared__ int   ix[2048];
  int t = threadIdx.x;
  int blk = blockIdx.x, which = blk >> 3, b = blk & 7;
  const float* s = w.sig + (long)which*BB*NN + (long)b*NN;
  for (int k=t;k<2048;k+=256) { v[k] = (k<NN)? s[k] : NEG_INF; ix[k] = (k<NN)? k : 0; }
  __syncthreads();
  for (int k2=2;k2<=2048;k2<<=1) {
    for (int j=k2>>1;j>0;j>>=1) {
      for (int i2=t;i2<2048;i2+=256) {
        int l = i2 ^ j;
        if (l > i2) {
          float a=v[i2], c=v[l];
          bool up = ((i2 & k2)==0);            // descending overall
          if (up ? (a<c) : (a>c)) { v[i2]=c; v[l]=a; int tm=ix[i2]; ix[i2]=ix[l]; ix[l]=tm; }
        }
      }
      __syncthreads();
    }
  }
  int*   idx = which ? w.idxT   : w.idxS;
  float* sel = which ? w.tgtSel : w.srcSel;
  const float* pts = which ? in.tgt : in.src;
  if (t < PP) {
    int id = ix[t];
    idx[b*PP+t] = id;
    for (int c=0;c<3;++c) {
      float pv = pts[((long)b*3+c)*NN + id];
      sel[((long)b*3+c)*PP + t] = pv;
      if (!which) w.srcCur[((long)b*3+c)*PP + t] = pv;
    }
  }
}

// ---------------- separable halves of the sf conv: A''[b,32,P], C'[b,32,P] ----------------
__global__ __launch_bounds__(256) void k_AC(In in, Ws w) {
  long gid = (long)blockIdx.x*256 + threadIdx.x;   // < 2*B*32*P
  int which = (int)(gid / (BB*32*PP));
  int rem   = (int)(gid % (BB*32*PP));
  int b = rem / (32*PP), o = (rem / PP) & 31, i = rem % PP;
  if (which == 0) {
    int id = w.idxS[b*PP+i];
    float acc = w.sfSh[o];
    for (int k=0;k<64;++k) acc += w.sfWa[o*64+k] * in.srcF[((long)b*64+k)*NN + id];
    w.A2[((long)b*32+o)*PP + i] = acc;
  } else {
    int id = w.idxT[b*PP+i];
    float acc = 0.f;
    for (int k=0;k<64;++k) acc += w.sfWb[o*64+k] * in.tgtF[((long)b*64+k)*NN + id];
    w.C2[((long)b*32+o)*PP + i] = acc;
  }
}

__global__ void k_init(Ws w) {
  int t = threadIdx.x;
  if (t < BB*9) w.Racc[t] = ((t % 9) % 4 == 0) ? 1.f : 0.f;
  if (t < BB*3) w.tacc[t] = 0.f;
}

// ---------------- main iteration kernel: WMMA over (i,j) pairs ----------------
// Each lane computes ONLY the 16 sim_feat channels of its B-fragment half:
// lanes 0-15 -> K=0..15, lanes 16-31 -> K=16..31 (no runtime register indexing).
// The c-loop is deliberately NOT unrolled to keep live ranges (and VGPR count)
// below 256 and avoid VGPR-MSB addressing overhead.
__global__ __launch_bounds__(256) void k_iter(Ws w) {
  __shared__ float lC[32*PP];     // C'[ch][j]
  __shared__ float lQ[3*PP];      // tgt positions
  __shared__ float lA[32*16];     // A''[ch][i_local] (shift folded in)
  __shared__ float lP[3*16];      // src positions (current)
  __shared__ float lWg[32*4];
  __shared__ float lW2[32], lSh[32];
  __shared__ float lW1[32*32];    // folded smW1
  __shared__ float redM[8*16];
  __shared__ int   redJ[8*16];
  __shared__ float redS[8*16*32];

  int t = threadIdx.x;
  int b = blockIdx.x >> 4, itile = blockIdx.x & 15;
  const float* C2 = w.C2 + (long)b*32*PP;
  const float* A2 = w.A2 + (long)b*32*PP;
  for (int k=t;k<32*PP;k+=256) lC[k] = C2[k];
  for (int k=t;k<3*PP;k+=256)  lQ[k] = w.tgtSel[(long)b*3*PP + k];
  for (int k=t;k<32*16;k+=256) { int ch=k>>4, ii=k&15; lA[k] = A2[ch*PP + itile*16 + ii]; }
  if (t < 48) { int c=t/16, ii=t%16; lP[t] = w.srcCur[((long)b*3+c)*PP + itile*16 + ii]; }
  if (t < 128) lWg[t] = w.sfWg[t];
  if (t < 32)  { lW2[t]=w.smW2v[t]; lSh[t]=w.smSh[t]; }
  for (int k=t;k<1024;k+=256) lW1[k] = w.smW1f[k];
  __syncthreads();
  float bsm2 = w.smB2v[0];

  int lane = t & 31, wid = t >> 5;
  int n = lane & 15, grp = lane >> 4;
  int chBase = grp << 4;          // this lane's channel half

  // A fragments for folded W_sm1 (rows 0-15 -> aW0, rows 16-31 -> aW1)
  // 16-bit A 16x32 layout (contiguous, matches B): g0: K=h ; g1: K=h+16
  v16h aW0, aW1;
  #pragma unroll
  for (int h=0; h<16; ++h) {
    int K = chBase + h;
    aW0[h] = (_Float16)lW1[n*32 + K];
    aW1[h] = (_Float16)lW1[(16+n)*32 + K];
  }

  for (int i=0; i<16; ++i) {
    float px=lP[0*16+i], py=lP[1*16+i], pz=lP[2*16+i];
    float bestM = NEG_INF; int bestJ = 0x7fffffff;
    float smax[16];
    #pragma unroll
    for (int h=0;h<16;++h) smax[h] = NEG_INF;

    #pragma unroll 1
    for (int c=0;c<2;++c) {
      int j = wid*32 + c*16 + n;
      float qx=lQ[0*PP+j], qy=lQ[1*PP+j], qz=lQ[2*PP+j];
      float dx=px-qx, dy=py-qy, dz=pz-qz;
      float dist = sqrtf(dx*dx + dy*dy + dz*dz);
      float inv  = 1.0f/(dist + EPSV);
      float ex=dx*inv, ey=dy*inv, ez=dz*inv;

      v16h bS;
      #pragma unroll
      for (int h=0; h<16; ++h) {
        int ch = chBase + h;
        float a = lA[ch*16+i] + lC[ch*PP+j]
                + lWg[ch*4+0]*dist + lWg[ch*4+1]*ex + lWg[ch*4+2]*ey + lWg[ch*4+3]*ez;
        a = fmaxf(a, 0.f);
        smax[h] = fmaxf(smax[h], a);
        bS[h] = (_Float16)a;
      }

      v8f d0 = {}; v8f d1 = {};
      d0 = __builtin_amdgcn_wmma_f32_16x16x32_f16(false, aW0, false, bS, (short)0, d0, false, false);
      d1 = __builtin_amdgcn_wmma_f32_16x16x32_f16(false, aW1, false, bS, (short)0, d1, false, false);

      float part = 0.f;
      #pragma unroll
      for (int r=0;r<8;++r) {
        int ch0 = r + (grp ? 8 : 0);  // LDS-indexed, no register select trees
        int ch1 = 16 + ch0;
        part += lW2[ch0]*fmaxf(d0[r] + lSh[ch0], 0.f);
        part += lW2[ch1]*fmaxf(d1[r] + lSh[ch1], 0.f);
      }
      float m = part + __shfl_xor(part, 16, 32) + bsm2;
      m = fminf(fmaxf(m, -20.f), 20.f);
      if (m > bestM || (m == bestM && j < bestJ)) { bestM = m; bestJ = j; }
    }
    // reduce across the 16 pair-lanes within each half
    #pragma unroll
    for (int off=1; off<16; off<<=1) {
      float om = __shfl_xor(bestM, off, 32);
      int   oj = __shfl_xor(bestJ, off, 32);
      if (om > bestM || (om == bestM && oj < bestJ)) { bestM = om; bestJ = oj; }
      #pragma unroll
      for (int h=0;h<16;++h) smax[h] = fmaxf(smax[h], __shfl_xor(smax[h], off, 32));
    }
    if (lane == 0) {
      redM[wid*16+i] = bestM; redJ[wid*16+i] = bestJ;
      #pragma unroll
      for (int h=0;h<16;++h) redS[(wid*16+i)*32+h] = smax[h];        // ch 0..15
    }
    if (lane == 16) {
      #pragma unroll
      for (int h=0;h<16;++h) redS[(wid*16+i)*32+16+h] = smax[h];     // ch 16..31
    }
  }
  __syncthreads();
  if (t < 16) {
    float bm = NEG_INF; int bj = 0x7fffffff;
    for (int wv=0; wv<8; ++wv) {
      float m2 = redM[wv*16+t]; int j2 = redJ[wv*16+t];
      if (m2 > bm || (m2 == bm && j2 < bj)) { bm = m2; bj = j2; }
    }
    w.corr[(long)b*PP + itile*16 + t] = bj;
  }
  for (int k=t;k<512;k+=256) {
    int ch = k & 31, ii = k >> 5;
    float mx = NEG_INF;
    for (int wv=0; wv<8; ++wv) mx = fmaxf(mx, redS[(wv*16+ii)*32+ch]);
    w.rowmax[((long)b*32+ch)*PP + itile*16 + ii] = mx;
  }
}

// ---------------- weight_fc on rowmax ----------------
__global__ __launch_bounds__(256) void k_weights(Ws w) {
  __shared__ float sW[32*32];
  __shared__ float sSh[32], sW2v[32];
  int t = threadIdx.x;
  for (int k=t;k<1024;k+=256) sW[k] = w.wfW1f[k];
  if (t<32) { sSh[t]=w.wfSh[t]; sW2v[t]=w.wfW2v[t]; }
  __syncthreads();
  int gid = blockIdx.x*256 + t;
  int b = gid / PP, p = gid % PP;
  float x[32];
  #pragma unroll
  for (int ch=0;ch<32;++ch) x[ch] = w.rowmax[((long)b*32+ch)*PP + p];
  float acc = w.wfB2v[0];
  for (int o=0;o<32;++o) {
    float h = sSh[o];
    #pragma unroll
    for (int ch=0;ch<32;++ch) h += sW[o*32+ch]*x[ch];
    acc += sW2v[o]*fmaxf(h, 0.f);
  }
  w.wts[gid] = acc;
}

__device__ inline float blockReduceSum(float v, float* red, int t) {
  red[t] = v; __syncthreads();
  for (int s=128; s>0; s>>=1) { if (t < s) red[t] += red[t+s]; __syncthreads(); }
  float r = red[0]; __syncthreads();
  return r;
}

// ---------------- weighted SVD + pose update, one block per batch ----------------
__global__ __launch_bounds__(256) void k_svd(Ws w) {
  __shared__ float srt[PP];
  __shared__ float red[PP];
  __shared__ float sR[9], sT[3];
  int b = blockIdx.x, t = threadIdx.x;

  float wv = 1.f/(1.f + expf(-w.wts[(long)b*PP + t]));
  srt[t] = wv;
  __syncthreads();
  // bitonic ascending sort for the median (torch lower-middle: index (P-1)/2)
  for (int k2=2; k2<=PP; k2<<=1) {
    for (int j=k2>>1; j>0; j>>=1) {
      int l = t ^ j;
      if (l > t) {
        float a=srt[t], c=srt[l];
        bool up = ((t & k2)==0);
        if (up ? (a>c) : (a<c)) { srt[t]=c; srt[l]=a; }
      }
      __syncthreads();
    }
  }
  float med = srt[(PP-1)/2];
  __syncthreads();
  wv = (wv >= med) ? wv : 0.f;
  float wsum = blockReduceSum(wv, red, t);
  float wn = wv / (wsum + EPSV);

  float p0 = w.srcCur[((long)b*3+0)*PP + t];
  float p1 = w.srcCur[((long)b*3+1)*PP + t];
  float p2 = w.srcCur[((long)b*3+2)*PP + t];
  int cj = w.corr[(long)b*PP + t];
  float q0 = w.tgtSel[((long)b*3+0)*PP + cj];
  float q1 = w.tgtSel[((long)b*3+1)*PP + cj];
  float q2 = w.tgtSel[((long)b*3+2)*PP + cj];

  const float invP = 1.f/(float)PP;
  float ms0 = blockReduceSum(p0, red, t)*invP;
  float ms1 = blockReduceSum(p1, red, t)*invP;
  float ms2 = blockReduceSum(p2, red, t)*invP;
  float mc0 = blockReduceSum(q0, red, t)*invP;
  float mc1 = blockReduceSum(q1, red, t)*invP;
  float mc2 = blockReduceSum(q2, red, t)*invP;
  float sc0=p0-ms0, sc1=p1-ms1, sc2=p2-ms2;
  float tc0=q0-mc0, tc1=q1-mc1, tc2=q2-mc2;

  float Hm[9];
  Hm[0]=blockReduceSum(wn*sc0*tc0,red,t); Hm[1]=blockReduceSum(wn*sc0*tc1,red,t); Hm[2]=blockReduceSum(wn*sc0*tc2,red,t);
  Hm[3]=blockReduceSum(wn*sc1*tc0,red,t); Hm[4]=blockReduceSum(wn*sc1*tc1,red,t); Hm[5]=blockReduceSum(wn*sc1*tc2,red,t);
  Hm[6]=blockReduceSum(wn*sc2*tc0,red,t); Hm[7]=blockReduceSum(wn*sc2*tc1,red,t); Hm[8]=blockReduceSum(wn*sc2*tc2,red,t);
  float tws0=blockReduceSum(wn*p0,red,t), tws1=blockReduceSum(wn*p1,red,t), tws2=blockReduceSum(wn*p2,red,t);
  float twc0=blockReduceSum(wn*q0,red,t), twc1=blockReduceSum(wn*q1,red,t), twc2=blockReduceSum(wn*q2,red,t);

  if (t == 0) {
    float A[9];
    for (int i=0;i<3;++i) for (int j=0;j<3;++j)
      A[i*3+j] = Hm[0*3+i]*Hm[0*3+j] + Hm[1*3+i]*Hm[1*3+j] + Hm[2*3+i]*Hm[2*3+j];
    float V[9] = {1,0,0, 0,1,0, 0,0,1};
    for (int sweep=0; sweep<16; ++sweep) {
      const int prs[3][2] = {{0,1},{0,2},{1,2}};
      for (int pi=0; pi<3; ++pi) {
        int p=prs[pi][0], q=prs[pi][1];
        float apq = A[p*3+q];
        if (fabsf(apq) < 1e-20f) continue;
        float theta = (A[q*3+q]-A[p*3+p])/(2.f*apq);
        float tt = (theta>=0.f?1.f:-1.f)/(fabsf(theta)+sqrtf(theta*theta+1.f));
        float c = rsqrtf(tt*tt+1.f), s = tt*c;
        float app=A[p*3+p], aqq=A[q*3+q];
        A[p*3+p] = app - tt*apq;
        A[q*3+q] = aqq + tt*apq;
        A[p*3+q] = 0.f; A[q*3+p] = 0.f;
        for (int k=0;k<3;++k) {
          if (k==p || k==q) continue;
          float akp=A[k*3+p], akq=A[k*3+q];
          A[k*3+p] = c*akp - s*akq; A[p*3+k] = A[k*3+p];
          A[k*3+q] = s*akp + c*akq; A[q*3+k] = A[k*3+q];
        }
        for (int k=0;k<3;++k) {
          float vkp=V[k*3+p], vkq=V[k*3+q];
          V[k*3+p] = c*vkp - s*vkq;
          V[k*3+q] = s*vkp + c*vkq;
        }
      }
    }
    float e[3] = {A[0], A[4], A[8]};
    for (int a2=0;a2<2;++a2) for (int b2=0;b2<2-a2;++b2) {
      if (e[b2] < e[b2+1]) {
        float tm=e[b2]; e[b2]=e[b2+1]; e[b2+1]=tm;
        for (int k=0;k<3;++k) { float tv2=V[k*3+b2]; V[k*3+b2]=V[k*3+b2+1]; V[k*3+b2+1]=tv2; }
      }
    }
    float S[3], U[9];
    for (int k=0;k<3;++k) {
      S[k] = sqrtf(fmaxf(e[k], 0.f));
      float u0 = Hm[0]*V[0*3+k] + Hm[1]*V[1*3+k] + Hm[2]*V[2*3+k];
      float u1 = Hm[3]*V[0*3+k] + Hm[4]*V[1*3+k] + Hm[5]*V[2*3+k];
      float u2 = Hm[6]*V[0*3+k] + Hm[7]*V[1*3+k] + Hm[8]*V[2*3+k];
      float inv = (S[k] > 1e-12f) ? 1.f/S[k] : 0.f;
      U[0*3+k]=u0*inv; U[1*3+k]=u1*inv; U[2*3+k]=u2*inv;
    }
    if (S[2] <= 1e-12f) {
      U[0*3+2] = U[1*3+0]*U[2*3+1] - U[2*3+0]*U[1*3+1];
      U[1*3+2] = U[2*3+0]*U[0*3+1] - U[0*3+0]*U[2*3+1];
      U[2*3+2] = U[0*3+0]*U[1*3+1] - U[1*3+0]*U[0*3+1];
    }
    float detU = U[0]*(U[4]*U[8]-U[5]*U[7]) - U[1]*(U[3]*U[8]-U[5]*U[6]) + U[2]*(U[3]*U[7]-U[4]*U[6]);
    float detV = V[0]*(V[4]*V[8]-V[5]*V[7]) - V[1]*(V[3]*V[8]-V[5]*V[6]) + V[2]*(V[3]*V[7]-V[4]*V[6]);
    float sg = (detU*detV < 0.f) ? -1.f : 1.f;
    float R[9];
    for (int i=0;i<3;++i) for (int j=0;j<3;++j)
      R[i*3+j] = V[i*3+0]*U[j*3+0] + V[i*3+1]*U[j*3+1] + sg*V[i*3+2]*U[j*3+2];
    float tv[3];
    tv[0] = -(R[0]*tws0 + R[1]*tws1 + R[2]*tws2) + twc0;
    tv[1] = -(R[3]*tws0 + R[4]*tws1 + R[5]*tws2) + twc1;
    tv[2] = -(R[6]*tws0 + R[7]*tws1 + R[8]*tws2) + twc2;
    for (int k=0;k<9;++k) sR[k]=R[k];
    for (int k=0;k<3;++k) sT[k]=tv[k];
    float Ro[9], Rn[9], to[3], tn[3];
    for (int k=0;k<9;++k) Ro[k] = w.Racc[b*9+k];
    for (int k=0;k<3;++k) to[k] = w.tacc[b*3+k];
    for (int i=0;i<3;++i) for (int j=0;j<3;++j)
      Rn[i*3+j] = R[i*3+0]*Ro[0*3+j] + R[i*3+1]*Ro[1*3+j] + R[i*3+2]*Ro[2*3+j];
    for (int i=0;i<3;++i)
      tn[i] = R[i*3+0]*to[0] + R[i*3+1]*to[1] + R[i*3+2]*to[2] + tv[i];
    for (int k=0;k<9;++k) w.Racc[b*9+k] = Rn[k];
    for (int k=0;k<3;++k) w.tacc[b*3+k] = tn[k];
  }
  __syncthreads();
  float np0 = sR[0]*p0 + sR[1]*p1 + sR[2]*p2 + sT[0];
  float np1 = sR[3]*p0 + sR[4]*p1 + sR[5]*p2 + sT[1];
  float np2 = sR[6]*p0 + sR[7]*p1 + sR[8]*p2 + sT[2];
  w.srcCur[((long)b*3+0)*PP + t] = np0;
  w.srcCur[((long)b*3+1)*PP + t] = np1;
  w.srcCur[((long)b*3+2)*PP + t] = np2;
}

__global__ void k_out(Ws w, float* out) {
  int t = threadIdx.x;
  if (t < BB*9) out[t] = w.Racc[t];
  if (t < BB*3) out[BB*9 + t] = w.tacc[t];
}

// ---------------- host launcher ----------------
extern "C" void kernel_launch(void* const* d_in, const int* in_sizes, int n_in,
                              void* d_out, int out_size, void* d_ws, size_t ws_size,
                              hipStream_t stream) {
  (void)in_sizes; (void)n_in; (void)out_size; (void)ws_size;
  In in;
  in.src  = (const float*)d_in[0];
  in.tgt  = (const float*)d_in[1];
  in.srcF = (const float*)d_in[2];
  in.tgtF = (const float*)d_in[3];
  in.sfW1=(const float*)d_in[4];  in.sfW2=(const float*)d_in[5];  in.sfB2=(const float*)d_in[6];
  in.sfG =(const float*)d_in[7];  in.sfBb=(const float*)d_in[8];  in.sfM =(const float*)d_in[9];  in.sfV=(const float*)d_in[10];
  in.sgW1=(const float*)d_in[11]; in.sgW2=(const float*)d_in[12]; in.sgW3=(const float*)d_in[13]; in.sgB3=(const float*)d_in[14];
  in.sg1G=(const float*)d_in[15]; in.sg1B=(const float*)d_in[16]; in.sg1M=(const float*)d_in[17]; in.sg1V=(const float*)d_in[18];
  in.sg2G=(const float*)d_in[19]; in.sg2B=(const float*)d_in[20]; in.sg2M=(const float*)d_in[21]; in.sg2V=(const float*)d_in[22];
  in.smW1=(const float*)d_in[23]; in.smW2=(const float*)d_in[24]; in.smB2=(const float*)d_in[25];
  in.smG =(const float*)d_in[26]; in.smBb=(const float*)d_in[27]; in.smM =(const float*)d_in[28]; in.smV=(const float*)d_in[29];
  in.wfW1=(const float*)d_in[30]; in.wfW2=(const float*)d_in[31]; in.wfB2=(const float*)d_in[32];
  in.wfG =(const float*)d_in[33]; in.wfBb=(const float*)d_in[34]; in.wfM =(const float*)d_in[35]; in.wfV=(const float*)d_in[36];

  char* base = (char*)d_ws; size_t off = 0;
  auto alloc = [&](size_t nElems) -> void* {
    void* r = base + off;
    off = (off + nElems*4 + 255) & ~(size_t)255;
    return r;
  };
  Ws w;
  w.sig   =(float*)alloc(2*BB*NN);
  w.idxS  =(int*)  alloc(BB*PP);   w.idxT=(int*)alloc(BB*PP);
  w.srcSel=(float*)alloc(BB*3*PP); w.tgtSel=(float*)alloc(BB*3*PP); w.srcCur=(float*)alloc(BB*3*PP);
  w.A2    =(float*)alloc(BB*32*PP); w.C2=(float*)alloc(BB*32*PP);   w.rowmax=(float*)alloc(BB*32*PP);
  w.wts   =(float*)alloc(BB*PP);   w.corr=(int*)alloc(BB*PP);
  w.Racc  =(float*)alloc(BB*9);    w.tacc=(float*)alloc(BB*3);
  w.sigW1 =(float*)alloc(64*64);   w.sigB1=(float*)alloc(64);
  w.sigW2 =(float*)alloc(32*64);   w.sigB2=(float*)alloc(32);
  w.sigW3 =(float*)alloc(32);      w.sigB3=(float*)alloc(1);
  w.sfWa  =(float*)alloc(32*64);   w.sfWb=(float*)alloc(32*64);
  w.sfWg  =(float*)alloc(32*4);    w.sfSh=(float*)alloc(32);
  w.smW1f =(float*)alloc(32*32);   w.smSh=(float*)alloc(32);
  w.smW2v =(float*)alloc(32);      w.smB2v=(float*)alloc(1);
  w.wfW1f =(float*)alloc(32*32);   w.wfSh=(float*)alloc(32);
  w.wfW2v =(float*)alloc(32);      w.wfB2v=(float*)alloc(1);

  k_prep<<<1,256,0,stream>>>(in, w);
  k_sig <<<(2*BB*NN)/256,256,0,stream>>>(in, w);
  k_topk<<<2*BB,256,0,stream>>>(in, w);
  k_AC  <<<(2*BB*32*PP)/256,256,0,stream>>>(in, w);
  k_init<<<1,128,0,stream>>>(w);
  for (int it=0; it<3; ++it) {
    k_iter   <<<BB*(PP/16),256,0,stream>>>(w);
    k_weights<<<(BB*PP)/256,256,0,stream>>>(w);
    k_svd    <<<BB,256,0,stream>>>(w);
  }
  k_out<<<1,128,0,stream>>>(w, (float*)d_out);
}
